// Attention_57346403336311
// MI455X (gfx1250) — matrix operand order repaired and checked by reading, not compiled, not run on hardware
//
#include <hip/hip_runtime.h>
#include <hip/hip_bf16.h>
#include <math.h>

typedef _Float16 f16;
typedef __attribute__((ext_vector_type(16))) _Float16 v16h;
typedef __attribute__((ext_vector_type(8)))  _Float16 v8h;
typedef __attribute__((ext_vector_type(8)))  float    v8f;
typedef __attribute__((ext_vector_type(4)))  int      i32x4;

#define HEADS  8
#define DH     64
#define BATCH  4
#define NSEQ   2048
#define DMODEL 512
#define NROWS  (BATCH*NSEQ)      // 8192
#define NQKV   (3*HEADS*DH)      // 1536
#define CENTER 32.0f             // sqrt(2*512): mean of the cdist outputs

// ---------------- helpers ----------------
__device__ __forceinline__ v8f vzero8() {
  v8f z;
#pragma unroll
  for (int i = 0; i < 8; i++) z[i] = 0.0f;
  return z;
}
__device__ __forceinline__ v16h comb(v8h lo, v8h hi) {
  return __builtin_shufflevector(lo, hi, 0,1,2,3,4,5,6,7,8,9,10,11,12,13,14,15);
}
__device__ __forceinline__ float redmax16(float v) {
#pragma unroll
  for (int m = 1; m < 16; m <<= 1) v = fmaxf(v, __shfl_xor(v, m, 32));
  return v;
}
__device__ __forceinline__ float redsum16(float v) {
#pragma unroll
  for (int m = 1; m < 16; m <<= 1) v += __shfl_xor(v, m, 32);
  return v;
}
__device__ __forceinline__ float redsum32(float v) {
#pragma unroll
  for (int m = 1; m < 32; m <<= 1) v += __shfl_xor(v, m, 32);
  return v;
}

// ---- CDNA5 async global->LDS (ASYNCcnt) with sync fallback ----
// Probe-learned signature: (i32x4 AS(1)*, i32x4 AS(3)*, imm offset, imm cpol);
// AS(1) prints as '__device__' in hipcc diagnostics.
#if __has_builtin(__builtin_amdgcn_global_load_async_to_lds_b128)
#define ASYNC_LDS 1
typedef __attribute__((address_space(1))) i32x4* as1_i32x4_ptr;
typedef __attribute__((address_space(3))) i32x4* as3_i32x4_ptr;
#else
#define ASYNC_LDS 0
#endif

__device__ __forceinline__ void copy16_g2l(const f16* g, f16* l) {
#if ASYNC_LDS
  __builtin_amdgcn_global_load_async_to_lds_b128((as1_i32x4_ptr)g,
                                                 (as3_i32x4_ptr)l, 0, 0);
#else
  *(uint4*)l = *(const uint4*)g;
#endif
}
__device__ __forceinline__ void async_wait() {
#if ASYNC_LDS && __has_builtin(__builtin_amdgcn_s_wait_asynccnt)
  __builtin_amdgcn_s_wait_asynccnt(0);
#endif
}

// A-operand fragment (16x32 f16): lane m=lane&15, half=lane>>4.
// elements 0..7  <- K = half*8 + 0..7        (16 contiguous bytes)
// elements 8..15 <- K = 16 + half*8 + 0..7   (16 contiguous bytes)
__device__ __forceinline__ v16h load_afrag(const f16* rowbase, int half) {
  v8h lo = *(const v8h*)(rowbase + half * 8);
  v8h hi = *(const v8h*)(rowbase + 16 + half * 8);
  return comb(lo, hi);
}
// B-operand fragment (32x16 f16, stored as Bt[N][K] row-major): lane n=lane&15.
// elements 0..7 <- K = half*8 + 0..7; elements 8..15 <- K = 16 + half*8 + 0..7
__device__ __forceinline__ v16h load_bfrag(const f16* rowbase, int half) {
  v8h lo = *(const v8h*)(rowbase + half * 8);
  v8h hi = *(const v8h*)(rowbase + 16 + half * 8);
  return comb(lo, hi);
}

// ---------------- prep: f32 -> f16, row stats ----------------
// mode 0: stat = sum(v^2)                 (x, w_qkv)
// mode 1: stat = sum(v^2) - 64*sum(v)     (w_out folded constant c_j)
__global__ __launch_bounds__(256) void prep_kernel(const float* __restrict__ src,
                                                   f16* __restrict__ dst,
                                                   float* __restrict__ stat,
                                                   int mode) {
  int row  = blockIdx.x * 8 + (threadIdx.x >> 5);
  int lane = threadIdx.x & 31;
  const float* p = src + (size_t)row * DMODEL;
  f16* q = dst + (size_t)row * DMODEL;
  float ssq = 0.f, ssum = 0.f;
#pragma unroll
  for (int i = 0; i < 16; i++) {
    float v = p[i * 32 + lane];
    q[i * 32 + lane] = (f16)v;
    ssq += v * v;
    ssum += v;
  }
  ssq  = redsum32(ssq);
  ssum = redsum32(ssum);
  if (lane == 0) stat[row] = mode ? (ssq - 2.0f * CENTER * ssum) : ssq;
}

// ---------------- row sum-of-squares over 64 f16 (q', k' norms) ----------------
__global__ __launch_bounds__(256) void rowsq_kernel(const f16* __restrict__ qh,
                                                    const f16* __restrict__ kh,
                                                    float* __restrict__ qs,
                                                    float* __restrict__ ks) {
  int row  = blockIdx.x * 8 + (threadIdx.x >> 5);
  int lane = threadIdx.x & 31;
  const f16* src = blockIdx.y ? kh : qh;
  float* dst     = blockIdx.y ? ks : qs;
  const f16* p = src + (size_t)row * DH + lane * 2;
  float a = (float)p[0], b = (float)p[1];
  float s = redsum32(a * a + b * b);
  if (lane == 0) dst[row] = s;
}

__global__ __launch_bounds__(256) void zero_kernel(float* __restrict__ p, int n) {
  int i = blockIdx.x * 256 + threadIdx.x;
  if (i < n) p[i] = 0.0f;
}

// ---------------- tiled GEMM: C = A[M,K] * Bt[N,K]^T, f16 WMMA, f32 acc ----
// 128x128 block, 8 waves, each wave 32(M) x 64(N). LDS rows padded to 40 f16.
// EPI 0: cdist-QKV epilogue -> centered q/k f16 [BH,N,64] and v^T f16 [BH,64,N]
// EPI 1: cdist-out epilogue -> f32 d_out
template <int EPI>
__global__ __launch_bounds__(256) void gemm_kernel(
    const f16* __restrict__ A, const f16* __restrict__ Bt,
    int M, int N, int K,
    const float* __restrict__ rowc, const float* __restrict__ colc,
    f16* __restrict__ qh, f16* __restrict__ kh, f16* __restrict__ vth,
    float* __restrict__ outp) {
  __shared__ __align__(16) f16 As[128 * 40];
  __shared__ __align__(16) f16 Bs[128 * 40];

  const int tid  = threadIdx.x;
  const int wave = tid >> 5, lane = tid & 31;
  const int lm   = lane & 15, half = lane >> 4;
  const int wm   = wave >> 1, wn = wave & 1;
  const int m0   = blockIdx.y * 128;
  const int n0   = blockIdx.x * 128;

  v8f acc[2][4];
#pragma unroll
  for (int i = 0; i < 2; i++)
#pragma unroll
    for (int j = 0; j < 4; j++) acc[i][j] = vzero8();

  for (int kb = 0; kb < K; kb += 32) {
    // cooperative stage: 128 rows x 32 f16 of A and Bt (16B chunks per lane)
#pragma unroll
    for (int c = 0; c < 2; c++) {
      int chunk = tid * 2 + c;        // 0..511
      int r = chunk >> 2, p = chunk & 3;
      copy16_g2l(&A[(size_t)(m0 + r) * K + kb + p * 8],  &As[r * 40 + p * 8]);
      copy16_g2l(&Bt[(size_t)(n0 + r) * K + kb + p * 8], &Bs[r * 40 + p * 8]);
    }
    if (kb + 32 < K) {  // prefetch next K-slab (global_prefetch_b8)
      __builtin_prefetch(&A[(size_t)(m0 + (tid >> 1)) * K + kb + 32], 0, 1);
      __builtin_prefetch(&Bt[(size_t)(n0 + (tid >> 1)) * K + kb + 32], 0, 1);
    }
    async_wait();
    __syncthreads();

    v16h af[2];
#pragma unroll
    for (int mt = 0; mt < 2; mt++)
      af[mt] = load_afrag(&As[(wm * 32 + mt * 16 + lm) * 40], half);
#pragma unroll
    for (int nt = 0; nt < 4; nt++) {
      v16h bf = load_bfrag(&Bs[(wn * 64 + nt * 16 + lm) * 40], half);
#pragma unroll
      for (int mt = 0; mt < 2; mt++)
        acc[mt][nt] = __builtin_amdgcn_wmma_f32_16x16x32_f16(
            false, af[mt], false, bf, (short)0, acc[mt][nt], false, false);
    }
    __syncthreads();
  }

  // epilogue: C/D layout: VGPR r, row m = r + 8*half, col n = lane&15
#pragma unroll
  for (int mt = 0; mt < 2; mt++) {
    int ibase = m0 + wm * 32 + mt * 16 + half * 8;
#pragma unroll
    for (int nt = 0; nt < 4; nt++) {
      int gj = n0 + wn * 64 + nt * 16 + lm;
      float cc = colc[gj];
      if (EPI == 0) {
        int sel = gj >> 9, h = (gj >> 6) & 7, d = gj & 63;
#pragma unroll
        for (int r = 0; r < 8; r++) {
          int i = ibase + r;
          float sq  = rowc[i] + cc - 2.0f * acc[mt][nt][r];
          float val = sqrtf(fmaxf(sq, 0.0f)) - CENTER;   // centered
          int b = i >> 11, ns = i & 2047;
          size_t bh = (size_t)(b * HEADS + h);
          if (sel == 0)      qh[(bh * NSEQ + ns) * DH + d]  = (f16)val;
          else if (sel == 1) kh[(bh * NSEQ + ns) * DH + d]  = (f16)val;
          else               vth[(bh * DH + d) * NSEQ + ns] = (f16)val;
        }
      } else {
#pragma unroll
        for (int r = 0; r < 8; r++) {
          int i = ibase + r;
          float sq = rowc[i] + cc - 2.0f * acc[mt][nt][r];
          outp[(size_t)i * N + gj] = sqrtf(fmaxf(sq, 0.0f));
        }
      }
    }
  }
}

// ---------------- flash attention (centered L2 scores) ----------------
// grid: (NSEQ/128, BATCH*HEADS); 8 waves, wave = 16 query rows.
// Row-sum of P computed by an extra WMMA against a ones B-matrix (acc #4),
// so only the row-max needs cross-lane shuffles.
__global__ __launch_bounds__(256) void attn_kernel(
    const f16* __restrict__ qh, const f16* __restrict__ kh,
    const f16* __restrict__ vth, const float* __restrict__ qs,
    const float* __restrict__ ks, f16* __restrict__ och,
    float* __restrict__ out_sq) {
  __shared__ __align__(16) f16 Ks[32 * 72];   // [32 keys][64 d], padded
  __shared__ __align__(16) f16 Vs[64 * 40];   // [64 d][32 keys], padded
  __shared__ __align__(16) f16 Ps[8][16 * 40];// per-wave P tile [16][32], padded

  const int tid  = threadIdx.x;
  const int wave = tid >> 5, lane = tid & 31;
  const int lm   = lane & 15, half = lane >> 4;
  const int bh   = blockIdx.y;
  const int q0   = blockIdx.x * 128 + wave * 16;
  const float scale = 0.125f;  // DH^-0.5

  // Q fragments (A operand), d split in two 32-blocks
  const f16* qrow = qh + ((size_t)bh * NSEQ + q0 + lm) * DH;
  v16h aq[2];
#pragma unroll
  for (int db = 0; db < 2; db++) aq[db] = load_afrag(qrow + db * 32, half);

  float qsr[8];
#pragma unroll
  for (int r = 0; r < 8; r++) qsr[r] = qs[(size_t)bh * NSEQ + q0 + half * 8 + r];

  v16h ones;
#pragma unroll
  for (int i = 0; i < 16; i++) ones[i] = (f16)1.0f;

  float mrow[8];
  v8f accv[5];  // [0..3]: P@V' d-tiles, [4]: P@ones = row sums
#pragma unroll
  for (int r = 0; r < 8; r++) mrow[r] = -3.0e38f;
#pragma unroll
  for (int t = 0; t < 5; t++) accv[t] = vzero8();

  for (int kb = 0; kb < NSEQ; kb += 32) {
    {  // stage K tile: 32 rows x 64 f16
      int r = tid >> 3, p = tid & 7;
      copy16_g2l(&kh[((size_t)bh * NSEQ + kb + r) * DH + p * 8],
                 &Ks[r * 72 + p * 8]);
    }
    {  // stage V^T tile: 64 rows x 32 f16
      int r = tid >> 2, p = tid & 3;
      copy16_g2l(&vth[((size_t)bh * DH + r) * NSEQ + kb + p * 8],
                 &Vs[r * 40 + p * 8]);
    }
    if (kb + 32 < NSEQ) {  // prefetch next tiles
      __builtin_prefetch(&kh[((size_t)bh * NSEQ + kb + 32 + (tid >> 3)) * DH], 0, 1);
      __builtin_prefetch(&vth[((size_t)bh * DH + (tid >> 2)) * NSEQ + kb + 32], 0, 1);
    }
    async_wait();
    __syncthreads();

    float ksv0 = ks[(size_t)bh * NSEQ + kb + lm];
    float ksv1 = ks[(size_t)bh * NSEQ + kb + 16 + lm];
    float sims[2][8];
#pragma unroll
    for (int jt = 0; jt < 2; jt++) {
      v8f s = vzero8();
#pragma unroll
      for (int db = 0; db < 2; db++) {
        v16h bf = load_bfrag(&Ks[(jt * 16 + lm) * 72 + db * 32], half);
        s = __builtin_amdgcn_wmma_f32_16x16x32_f16(false, aq[db], false, bf,
                                                   (short)0, s, false, false);
      }
      float kv = jt ? ksv1 : ksv0;
#pragma unroll
      for (int r = 0; r < 8; r++)
        sims[jt][r] = (2.0f * s[r] - qsr[r] - kv) * scale;
    }

    // online softmax: row max via shuffles; row sum via WMMA ones-column
#pragma unroll
    for (int r = 0; r < 8; r++) {
      float mloc = redmax16(fmaxf(sims[0][r], sims[1][r]));
      float mnew = fmaxf(mrow[r], mloc);
      float corr = __expf(mrow[r] - mnew);
      float p0 = __expf(sims[0][r] - mnew);
      float p1 = __expf(sims[1][r] - mnew);
      mrow[r] = mnew;
#pragma unroll
      for (int t = 0; t < 5; t++) accv[t][r] = accv[t][r] * corr;
      int prow = r + half * 8;
      Ps[wave][prow * 40 + lm]      = (f16)p0;
      Ps[wave][prow * 40 + 16 + lm] = (f16)p1;
    }

    // P @ [V' | ones] : A = P (16x32)
    v16h pf = load_afrag(&Ps[wave][lm * 40], half);
#pragma unroll
    for (int nt = 0; nt < 4; nt++) {
      v16h vf = load_bfrag(&Vs[(nt * 16 + lm) * 40], half);
      accv[nt] = __builtin_amdgcn_wmma_f32_16x16x32_f16(false, pf, false, vf,
                                                        (short)0, accv[nt],
                                                        false, false);
    }
    accv[4] = __builtin_amdgcn_wmma_f32_16x16x32_f16(false, pf, false, ones,
                                                     (short)0, accv[4],
                                                     false, false);
    __syncthreads();
  }

  // epilogue: oc = (P@V')/rowsum  (== out - 32); out_sq += sum((oc+32)^2)
  const int b = bh >> 3, h = bh & 7;
#pragma unroll
  for (int r = 0; r < 8; r++) {
    int qg = q0 + half * 8 + r;
    float inv = 1.0f / accv[4][r];
    float part = 0.0f;
#pragma unroll
    for (int nt = 0; nt < 4; nt++) {
      float oc = accv[nt][r] * inv;
      int d = nt * 16 + lm;
      och[((size_t)b * NSEQ + qg) * DMODEL + h * DH + d] = (f16)oc;
      float ov = oc + CENTER;
      part += ov * ov;
    }
    part = redsum16(part);
    if (lm == 0) atomicAdd(&out_sq[b * NSEQ + qg], part);
  }
}

// ---------------- host ----------------
extern "C" void kernel_launch(void* const* d_in, const int* in_sizes, int n_in,
                              void* d_out, int out_size, void* d_ws,
                              size_t ws_size, hipStream_t stream) {
  (void)in_sizes; (void)n_in; (void)out_size; (void)ws_size;
  const float* x     = (const float*)d_in[0];
  const float* w_qkv = (const float*)d_in[1];
  const float* w_out = (const float*)d_in[2];
  float* out = (float*)d_out;

  char* w = (char*)d_ws;
  auto alloc = [&](size_t bytes) -> void* {
    void* p = (void*)w;
    w += (bytes + 255) & ~(size_t)255;
    return p;
  };
  f16*   xh     = (f16*)alloc((size_t)NROWS * DMODEL * 2);
  float* x_sq   = (float*)alloc((size_t)NROWS * 4);
  f16*   wqh    = (f16*)alloc((size_t)NQKV * DMODEL * 2);
  float* wq_sq  = (float*)alloc((size_t)NQKV * 4);
  f16*   woh    = (f16*)alloc((size_t)DMODEL * DMODEL * 2);
  float* cj     = (float*)alloc((size_t)DMODEL * 4);
  f16*   qh     = (f16*)alloc((size_t)NROWS * DMODEL * 2);
  f16*   kh     = (f16*)alloc((size_t)NROWS * DMODEL * 2);
  f16*   vth    = (f16*)alloc((size_t)NROWS * DMODEL * 2);
  float* qs     = (float*)alloc((size_t)BATCH * HEADS * NSEQ * 4);
  float* ks     = (float*)alloc((size_t)BATCH * HEADS * NSEQ * 4);
  f16*   och    = (f16*)alloc((size_t)NROWS * DMODEL * 2);
  float* out_sq = (float*)alloc((size_t)NROWS * 4);

  prep_kernel<<<NROWS / 8, 256, 0, stream>>>(x, xh, x_sq, 0);
  prep_kernel<<<NQKV / 8, 256, 0, stream>>>(w_qkv, wqh, wq_sq, 0);
  prep_kernel<<<DMODEL / 8, 256, 0, stream>>>(w_out, woh, cj, 1);

  gemm_kernel<0><<<dim3(NQKV / 128, NROWS / 128), 256, 0, stream>>>(
      xh, wqh, NROWS, NQKV, DMODEL, x_sq, wq_sq, qh, kh, vth, nullptr);

  rowsq_kernel<<<dim3(BATCH * HEADS * NSEQ / 8, 2), 256, 0, stream>>>(qh, kh, qs, ks);
  zero_kernel<<<NROWS / 256, 256, 0, stream>>>(out_sq, NROWS);

  attn_kernel<<<dim3(NSEQ / 128, BATCH * HEADS), 256, 0, stream>>>(
      qh, kh, vth, qs, ks, och, out_sq);

  gemm_kernel<1><<<dim3(DMODEL / 128, NROWS / 128), 256, 0, stream>>>(
      och, woh, NROWS, DMODEL, DMODEL, out_sq, cj, nullptr, nullptr, nullptr, out);
}
